// LinearQuantizer_26285199851693
// MI455X (gfx1250) — 8-bit matrix operand order repaired and checked by reading, compiled, not run on hardware
//
#include <hip/hip_runtime.h>
#include <hip/hip_bf16.h>

typedef __attribute__((ext_vector_type(8))) int v8i;

#define QMAX 127.0f
#define EPSQ 1e-8f

static constexpr int KDIM = 4096;
static constexpr int NDIM = 4096;
static constexpr int BM = 128, BN = 128, BK = 128;
static constexpr int LDSS = BK + 16;
static constexpr unsigned BUFSZ = (BM + BN) * LDSS;

__global__ void init_ws_kernel(unsigned* amax) { *amax = 0u; }

__global__ void absmax_x_kernel(const float4* __restrict__ x, unsigned* __restrict__ amax, int n4) {
  __shared__ float red[256];
  float m = 0.0f;
  for (int i = blockIdx.x * blockDim.x + threadIdx.x; i < n4; i += gridDim.x * blockDim.x) {
    float4 v = x[i];
    m = fmaxf(m, fmaxf(fmaxf(fabsf(v.x), fabsf(v.y)), fmaxf(fabsf(v.z), fabsf(v.w))));
  }
  red[threadIdx.x] = m;
  __syncthreads();
  for (int off = 128; off > 0; off >>= 1) {
    if (threadIdx.x < off) red[threadIdx.x] = fmaxf(red[threadIdx.x], red[threadIdx.x + off]);
    __syncthreads();
  }
  if (threadIdx.x == 0) atomicMax(amax, __float_as_uint(red[0]));
}

__device__ __forceinline__ int pack4(float4 v, float s) {
  int q0 = __float2int_rn(fminf(fmaxf(v.x / s, -QMAX), QMAX));
  int q1 = __float2int_rn(fminf(fmaxf(v.y / s, -QMAX), QMAX));
  int q2 = __float2int_rn(fminf(fmaxf(v.z / s, -QMAX), QMAX));
  int q3 = __float2int_rn(fminf(fmaxf(v.w / s, -QMAX), QMAX));
  return (q0 & 255) | ((q1 & 255) << 8) | ((q2 & 255) << 16) | (q3 << 24);
}

__global__ void quant_w_kernel(const float* __restrict__ w, signed char* __restrict__ qw,
                               float* __restrict__ wscale) {
  const int row = blockIdx.x;
  const float4* wr4 = (const float4*)(w + (size_t)row * KDIM);
  __shared__ float red[256];
  float4 vals[4];
  float m = 0.0f;
#pragma unroll
  for (int i = 0; i < 4; ++i) {
    vals[i] = wr4[threadIdx.x + i * 256];
    m = fmaxf(m, fmaxf(fmaxf(fabsf(vals[i].x), fabsf(vals[i].y)),
                       fmaxf(fabsf(vals[i].z), fabsf(vals[i].w))));
  }
  red[threadIdx.x] = m;
  __syncthreads();
  for (int off = 128; off > 0; off >>= 1) {
    if (threadIdx.x < off) red[threadIdx.x] = fmaxf(red[threadIdx.x], red[threadIdx.x + off]);
    __syncthreads();
  }
  const float s = fmaxf(red[0] / QMAX, EPSQ);
  if (threadIdx.x == 0) wscale[row] = s;
  int* qrow = (int*)(qw + (size_t)row * KDIM);
#pragma unroll
  for (int i = 0; i < 4; ++i) qrow[threadIdx.x + i * 256] = pack4(vals[i], s);
}

__global__ void quant_x_kernel(const float4* __restrict__ x, int* __restrict__ qx,
                               const unsigned* __restrict__ amax, int n4) {
  const float s = fmaxf(__uint_as_float(*amax) / QMAX, EPSQ);
  for (int i = blockIdx.x * blockDim.x + threadIdx.x; i < n4; i += gridDim.x * blockDim.x)
    qx[i] = pack4(x[i], s);
}

__device__ __forceinline__ void issue_tile_async(const signed char* gA, const signed char* gB,
                                                 int k0, unsigned bufBase, int tid) {
#pragma unroll
  for (int c = 0; c < 4; ++c) {
    const int chunk = tid + c * 256;
    const int row   = chunk >> 3;
    const int off   = (chunk & 7) << 4;
    const unsigned la = bufBase + (unsigned)(row * LDSS + off);
    const unsigned lb = la + (unsigned)(BM * LDSS);
    const signed char* pa = gA + (size_t)row * KDIM + k0 + off;
    const signed char* pb = gB + (size_t)row * KDIM + k0 + off;
    asm volatile("global_load_async_to_lds_b128 %0, %1, off" :: "v"(la), "v"(pa) : "memory");
    asm volatile("global_load_async_to_lds_b128 %0, %1, off" :: "v"(lb), "v"(pb) : "memory");
  }
}

__global__ __launch_bounds__(256) void gemm_i8_kernel(
    const signed char* __restrict__ qx, const signed char* __restrict__ qw,
    const float* __restrict__ wscale, const float* __restrict__ bias,
    const unsigned* __restrict__ amax, float* __restrict__ out) {
  __shared__ __align__(16) unsigned char sMem[2 * BUFSZ];

  const int tid   = threadIdx.x;
  const int lane  = tid & 31;
  const int wave  = tid >> 5;
  const int wm    = wave & 3;
  const int wn    = wave >> 2;
  const int lrow  = lane & 15;
  const int khalf = lane >> 4;

  const int m0 = blockIdx.y * BM;
  const int n0 = blockIdx.x * BN;

  const float a_scale = fmaxf(__uint_as_float(*amax) / QMAX, EPSQ);

  const signed char* gA = qx + (size_t)m0 * KDIM;
  const signed char* gB = qw + (size_t)n0 * KDIM;

  const unsigned ldsBase = (unsigned)(uintptr_t)(void*)&sMem[0];

  const v8i vzero = {0, 0, 0, 0, 0, 0, 0, 0};
  v8i acc[2][4];
#pragma unroll
  for (int mt = 0; mt < 2; ++mt)
#pragma unroll
    for (int nt = 0; nt < 4; ++nt) acc[mt][nt] = vzero;

  issue_tile_async(gA, gB, 0, ldsBase, tid);

  for (int k0 = 0; k0 < KDIM; k0 += BK) {
    const int cur = (k0 / BK) & 1;

    if (k0 + BK < KDIM) {
      issue_tile_async(gA, gB, k0 + BK, ldsBase + (unsigned)(cur ^ 1) * BUFSZ, tid);
      asm volatile("s_wait_asynccnt 0x8" ::: "memory");
    } else {
      asm volatile("s_wait_asynccnt 0x0" ::: "memory");
    }
    __syncthreads();

    if (k0 + 2 * BK < KDIM) {
      const int prow = tid >> 1;
      const int poff = (tid & 1) * 64;
      __builtin_prefetch(gA + (size_t)prow * KDIM + k0 + 2 * BK + poff, 0, 1);
      __builtin_prefetch(gB + (size_t)prow * KDIM + k0 + 2 * BK + poff, 0, 1);
    }

    const unsigned char* sA = &sMem[cur * BUFSZ];
    const unsigned char* sB = sA + BM * LDSS;

#pragma unroll
    for (int ks = 0; ks < BK; ks += 64) {
      v8i a[2];
#pragma unroll
      for (int mt = 0; mt < 2; ++mt) {
        const unsigned char* p = &sA[(wm * 32 + mt * 16 + lrow) * LDSS + ks + khalf * 16];
        int2 p0 = *(const int2*)(p + 0);
        int2 p1 = *(const int2*)(p + 8);
        int2 p2 = *(const int2*)(p + 32);
        int2 p3 = *(const int2*)(p + 40);
        v8i af = {p0.x, p0.y, p1.x, p1.y, p2.x, p2.y, p3.x, p3.y};
        a[mt] = af;
      }
      v8i b[4];
#pragma unroll
      for (int nt = 0; nt < 4; ++nt) {
        const unsigned char* p = &sB[(wn * 64 + nt * 16 + lrow) * LDSS + ks + khalf * 16];
        int4 q0 = *(const int4*)(p + 0);
        int4 q1 = *(const int4*)(p + 32);
        v8i bf = {q0.x, q0.y, q0.z, q0.w, q1.x, q1.y, q1.z, q1.w};
        b[nt] = bf;
      }
#pragma unroll
      for (int mt = 0; mt < 2; ++mt)
#pragma unroll
        for (int nt = 0; nt < 4; ++nt)
          acc[mt][nt] = __builtin_amdgcn_wmma_i32_16x16x64_iu8(
               true, a[mt],  true, b[nt], acc[mt][nt],
               false,  false);
    }
    __syncthreads();
  }

#pragma unroll
  for (int nt = 0; nt < 4; ++nt) {
    const int col = n0 + wn * 64 + nt * 16 + lrow;
    const float f = a_scale * wscale[col];
    const float bb = bias[col];
#pragma unroll
    for (int mt = 0; mt < 2; ++mt) {
      const int rbase = m0 + wm * 32 + mt * 16 + khalf * 8;
#pragma unroll
      for (int j = 0; j < 8; ++j)
        out[(size_t)(rbase + j) * NDIM + col] = (float)acc[mt][nt][j] * f + bb;
    }
  }
}

extern "C" void kernel_launch(void* const* d_in, const int* in_sizes, int n_in,
                              void* d_out, int out_size, void* d_ws, size_t ws_size,
                              hipStream_t stream) {
  const float* x    = (const float*)d_in[0];
  const float* w    = (const float*)d_in[1];
  const float* bias = (const float*)d_in[2];
  float* out        = (float*)d_out;

  const int M = in_sizes[0] / KDIM;

  const size_t OFF_WSCALE = 1024;
  const size_t OFF_QW     = 32768;
  const size_t OFF_QX     = OFF_QW + (size_t)NDIM * KDIM;
  const size_t NEEDED     = OFF_QX + (size_t)M * KDIM;
  if (ws_size < NEEDED) return;

  unsigned*    amax   = (unsigned*)d_ws;
  float*       wscale = (float*)((char*)d_ws + OFF_WSCALE);
  signed char* qw     = (signed char*)((char*)d_ws + OFF_QW);
  signed char* qx     = (signed char*)((char*)d_ws + OFF_QX);

  const int n4 = (M * KDIM) / 4;

  init_ws_kernel<<<1, 1, 0, stream>>>(amax);
  absmax_x_kernel<<<4096, 256, 0, stream>>>((const float4*)x, amax, n4);
  quant_w_kernel<<<NDIM, 256, 0, stream>>>(w, qw, wscale);
  quant_x_kernel<<<8192, 256, 0, stream>>>((const float4*)x, (int*)qx, amax, n4);

  dim3 grid(NDIM / BN, M / BM);
  gemm_i8_kernel<<<grid, 256, 0, stream>>>(qx, qw, wscale, bias, amax, out);
}
